// Freq_Self_Att_87488483819797
// MI455X (gfx1250) — compile-verified, matrix operand order checked by reading, not run on hardware
//
#include <hip/hip_runtime.h>

typedef __attribute__((ext_vector_type(16))) __bf16 v16bf;
typedef __attribute__((ext_vector_type(8)))  __bf16 v8bf;
typedef __attribute__((ext_vector_type(8)))  float  v8f;

#define B_   64
#define C_   1024
#define T_   1000
#define TP_  1024      // T padded to multiple of 64
#define D_   256
#define P_   100
#define SCALE_INV (1.0f / 16.0f)   // 1/sqrt(256)

// ---------------------------------------------------------------------------
// Kernel 1: input f32 [B][C][T]  ->  bf16 [B][TP][C]  (t-major for WMMA B-op)
// ---------------------------------------------------------------------------
__global__ __launch_bounds__(256)
void k_transpose(const float* __restrict__ in, __bf16* __restrict__ inT) {
  __shared__ __bf16 tile[32][33];
  int b  = blockIdx.z;
  int t0 = blockIdx.x * 32;
  int c0 = blockIdx.y * 32;
  int tx = threadIdx.x & 31;
  int ty = threadIdx.x >> 5;                 // 0..7
  const float* src = in + (size_t)b * C_ * T_;
  #pragma unroll
  for (int r = 0; r < 4; ++r) {
    int c = c0 + ty + r * 8;
    int t = t0 + tx;
    float v = (t < T_) ? src[(size_t)c * T_ + t] : 0.0f;
    tile[ty + r * 8][tx] = (__bf16)v;        // tile[c_local][t_local]
  }
  __syncthreads();
  __bf16* dst = inT + (size_t)b * TP_ * C_;
  #pragma unroll
  for (int r = 0; r < 4; ++r) {
    int tl = ty + r * 8;
    dst[(size_t)(t0 + tl) * C_ + (c0 + tx)] = tile[tx][tl];
  }
}

// ---------------------------------------------------------------------------
// Kernel 2: adaptive-avg-pool(100) + k/q projection (+bias), bf16 out
// one block per (b,c) row
// ---------------------------------------------------------------------------
__global__ __launch_bounds__(256)
void k_pool_kq(const float* __restrict__ in,
               const float* __restrict__ Wk, const float* __restrict__ bk,
               const float* __restrict__ Wq, const float* __restrict__ bq,
               __bf16* __restrict__ kB, __bf16* __restrict__ qB) {
  __shared__ float x[P_];
  int row = blockIdx.x;                      // b*C + c
  int tid = threadIdx.x;
  const float* src = in + (size_t)row * T_;
  if (tid < P_) {
    float s = 0.f;
    const float* p = src + tid * 10;
    #pragma unroll
    for (int i = 0; i < 10; ++i) s += p[i];
    x[tid] = s * 0.1f;
  }
  __syncthreads();
  int d = tid;                               // 0..255
  const float* wk = Wk + d * P_;
  const float* wq = Wq + d * P_;
  float ak = bk[d], aq = bq[d];
  #pragma unroll 4
  for (int p = 0; p < P_; ++p) {
    float xv = x[p];
    ak = fmaf(xv, wk[p], ak);
    aq = fmaf(xv, wq[p], aq);
  }
  kB[(size_t)row * D_ + d] = (__bf16)ak;
  qB[(size_t)row * D_ + d] = (__bf16)aq;
}

// ---------------------------------------------------------------------------
// Kernel 3: S = k·qT / 16, row softmax, write P (bf16)
// one block per (b, 16-row tile); 8 waves, each wave owns 8 column tiles
// ---------------------------------------------------------------------------
__global__ __launch_bounds__(256)
void k_att(const __bf16* __restrict__ kB, const __bf16* __restrict__ qB,
           __bf16* __restrict__ Pm) {
  __shared__ float S[16 * C_];               // 64 KB
  int blk  = blockIdx.x;
  int b    = blk >> 6;                       // / (C/16)
  int i0   = (blk & 63) << 4;
  int tid  = threadIdx.x;
  int wave = tid >> 5, lane = tid & 31;
  int hl    = lane & 15;
  int khalf = (lane >= 16) ? 8 : 0;          // ISA per-lane K swizzle

  // Preload A = k rows [i0..i0+15], lane m = hl, two b128 per K-step
  union { v16bf v; v8bf h[2]; } A[8];
  const __bf16* abase = kB + ((size_t)(b * C_) + i0 + hl) * D_ + khalf;
  #pragma unroll
  for (int kk = 0; kk < 8; ++kk) {
    A[kk].h[0] = *(const v8bf*)(abase + kk * 32);
    A[kk].h[1] = *(const v8bf*)(abase + kk * 32 + 16);
  }

  const __bf16* qbase0 = qB + ((size_t)(b * C_) + hl) * D_ + khalf;
  int mbase = (lane >= 16) ? 8 : 0;
  #pragma unroll 1
  for (int jj = 0; jj < 8; ++jj) {
    int j0 = (wave + jj * 8) << 4;
    const __bf16* bbase = qbase0 + (size_t)j0 * D_;
    v8f acc = {};
    #pragma unroll
    for (int kk = 0; kk < 8; ++kk) {
      union { v16bf v; v8bf h[2]; } Bt;
      Bt.h[0] = *(const v8bf*)(bbase + kk * 32);
      Bt.h[1] = *(const v8bf*)(bbase + kk * 32 + 16);
      acc = __builtin_amdgcn_wmma_f32_16x16x32_bf16(
          false, A[kk].v, false, Bt.v, (short)0, acc, false, false);
    }
    #pragma unroll
    for (int r = 0; r < 8; ++r)
      S[(mbase + r) * C_ + j0 + hl] = acc[r] * SCALE_INV;
  }
  __syncthreads();

  // Row softmax: wave handles rows {wave, wave+8}
  #pragma unroll 1
  for (int mi = 0; mi < 2; ++mi) {
    int m = wave + mi * 8;
    float* srow = &S[m * C_];
    float mx = -3.0e38f;
    for (int j = lane; j < C_; j += 32) mx = fmaxf(mx, srow[j]);
    #pragma unroll
    for (int o = 16; o > 0; o >>= 1) mx = fmaxf(mx, __shfl_xor(mx, o, 32));
    float sum = 0.f;
    for (int j = lane; j < C_; j += 32) {
      float e = __expf(srow[j] - mx);
      srow[j] = e;
      sum += e;
    }
    #pragma unroll
    for (int o = 16; o > 0; o >>= 1) sum += __shfl_xor(sum, o, 32);
    float inv = 1.0f / sum;
    __bf16* prow = Pm + ((size_t)(b * C_) + i0 + m) * C_;
    for (int j = lane; j < C_; j += 32)
      prow[j] = (__bf16)(srow[j] * inv);
  }
}

// ---------------------------------------------------------------------------
// Kernel 4: out = P @ input   (A = P bf16 [C][C], B = inT bf16 [TP][C])
// block = 64x256 output tile; 8 waves as 2(row)x4(col); wave = 32x64 tile
// (2 A-tiles x 4 B-tiles = 8 WMMA per K-step -> 1.5 b128 loads per WMMA)
// ---------------------------------------------------------------------------
__global__ __launch_bounds__(256)
void k_out(const __bf16* __restrict__ Pm, const __bf16* __restrict__ inT,
           float* __restrict__ out) {
  int b  = blockIdx.z;
  int i0 = blockIdx.y * 64;                  // C/64 = 16
  int t0 = blockIdx.x * 256;                 // TP/256 = 4
  int tid  = threadIdx.x;
  int wave = tid >> 5, lane = tid & 31;
  int hl    = lane & 15;
  int khalf = (lane >= 16) ? 8 : 0;
  int iw = (wave & 1) * 32;                  // 2 row groups of 32
  int tw = (wave >> 1) * 64;                 // 4 col groups of 64

  const __bf16* a0 = Pm + ((size_t)(b * C_) + i0 + iw + hl) * C_ + khalf;
  const __bf16* a1 = a0 + (size_t)16 * C_;
  const __bf16* bb = inT + ((size_t)b * TP_ + t0 + tw + hl) * C_ + khalf;

  v8f acc[2][4] = {};
  #pragma unroll 2
  for (int kk = 0; kk < 32; ++kk) {
    int off = kk * 32;
    union { v16bf v; v8bf h[2]; } A[2], Bt[4];
    A[0].h[0] = *(const v8bf*)(a0 + off);
    A[0].h[1] = *(const v8bf*)(a0 + off + 16);
    A[1].h[0] = *(const v8bf*)(a1 + off);
    A[1].h[1] = *(const v8bf*)(a1 + off + 16);
    #pragma unroll
    for (int n = 0; n < 4; ++n) {
      const __bf16* bp = bb + (size_t)(n * 16) * C_ + off;
      Bt[n].h[0] = *(const v8bf*)(bp);
      Bt[n].h[1] = *(const v8bf*)(bp + 16);
    }
    #pragma unroll
    for (int mi = 0; mi < 2; ++mi)
      #pragma unroll
      for (int n = 0; n < 4; ++n)
        acc[mi][n] = __builtin_amdgcn_wmma_f32_16x16x32_bf16(
            false, A[mi].v, false, Bt[n].v, (short)0, acc[mi][n], false, false);
  }

  int mbase = (lane >= 16) ? 8 : 0;
  #pragma unroll
  for (int mi = 0; mi < 2; ++mi) {
    float* obase = out + ((size_t)(b * C_) + i0 + iw + mi * 16 + mbase) * (size_t)T_;
    #pragma unroll
    for (int r = 0; r < 8; ++r) {
      float* orow = obase + (size_t)r * T_;
      #pragma unroll
      for (int n = 0; n < 4; ++n) {
        int t = t0 + tw + n * 16 + hl;
        if (t < T_) orow[t] = acc[mi][n][r];
      }
    }
  }
}

// ---------------------------------------------------------------------------
extern "C" void kernel_launch(void* const* d_in, const int* in_sizes, int n_in,
                              void* d_out, int out_size, void* d_ws, size_t ws_size,
                              hipStream_t stream) {
  const float* input = (const float*)d_in[0];
  const float* Wk    = (const float*)d_in[1];
  const float* bk    = (const float*)d_in[2];
  const float* Wq    = (const float*)d_in[3];
  const float* bq    = (const float*)d_in[4];
  float* out = (float*)d_out;

  // ws layout (all 16B-aligned): inT | kB | qB | P
  char* w = (char*)d_ws;
  __bf16* inT = (__bf16*)w;                                   // B*TP*C bf16
  __bf16* kB  = (__bf16*)(w + (size_t)B_ * TP_ * C_ * 2);     // B*C*D bf16
  __bf16* qB  = kB + (size_t)B_ * C_ * D_;                    // B*C*D bf16
  __bf16* Pm  = qB + (size_t)B_ * C_ * D_;                    // B*C*C bf16

  k_transpose<<<dim3(TP_ / 32, C_ / 32, B_), 256, 0, stream>>>(input, inT);
  k_pool_kq<<<dim3(B_ * C_), 256, 0, stream>>>(input, Wk, bk, Wq, bq, kB, qB);
  k_att<<<dim3(B_ * (C_ / 16)), 256, 0, stream>>>(kB, qB, Pm);
  k_out<<<dim3(TP_ / 256, C_ / 64, B_), 256, 0, stream>>>(Pm, inT, out);
}
